// Model_84911503442088
// MI455X (gfx1250) — matrix operand order repaired and checked by reading, not compiled, not run on hardware
//
#include <hip/hip_runtime.h>
#include <hip/hip_bf16.h>
#include <hip/hip_fp16.h>

// ---------------------------------------------------------------------------
// Graph TransformerConv (4 layers) for MI455X / gfx1250, wave32 + WMMA f16.
// ---------------------------------------------------------------------------

typedef __attribute__((ext_vector_type(16))) _Float16 v16h;
typedef __attribute__((ext_vector_type(8)))  float    v8f;

#define DEVI static __device__ __forceinline__

constexpr int B_   = 256;
constexpr int NPG  = 128;
constexpr int N_   = B_ * NPG;     // 32768
constexpr int C_   = 64;
constexpr int H_   = 4;
constexpr int L_   = 4;
constexpr int E_   = 294912;
constexpr int XD_  = 16;
constexpr int ED_  = 8;
constexpr int QKV  = H_ * C_;      // 256
constexpr int WCAT = 3 * QKV + C_; // 832 fused output cols: Q|K|V|skip
constexpr int ETILES = E_ / 16;    // 18432 edge tiles

// ---------------- WMMA helpers (contiguous k = 16h + s in A and B) ---------------

DEVI v8f wmma_f16(v16h a, v16h b, v8f c) {
  // v_wmma_f32_16x16x32_f16
  return __builtin_amdgcn_wmma_f32_16x16x32_f16(
      /*neg_a=*/false, a, /*neg_b=*/false, b,
      /*c_mod=*/(short)0, c, /*reuse_a=*/false, /*reuse_b=*/false);
}

// A-matrix 16x32 f16 tile: lane<16 holds M=lane, K=k0..k0+15;
// lane>=16 holds M=lane-16, K=k0+16..k0+31.
DEVI v16h load_a16x32(const _Float16* abase, int rowStride, int k0, int lane) {
  int m  = lane & 15;
  int ko = (lane >> 4) << 4;
  const _Float16* p = abase + (size_t)m * rowStride + k0 + ko;
  v16h a;
#pragma unroll
  for (int j = 0; j < 8; ++j) { a[j] = p[j]; a[8 + j] = p[8 + j]; }
  return a;
}

// B-matrix 32x16 f16 tile from a pre-transposed weight (Wt[outCol][64]):
// lane holds column n = n0+(lane&15), K = k0 + (lane>=16 ? 16 : 0) .. +15.
DEVI v16h load_b32x16(const _Float16* Wt, int n0, int k0, int lane) {
  int n  = n0 + (lane & 15);
  int ko = k0 + ((lane >> 4) << 4);
  const _Float16* p = Wt + (size_t)n * 64 + ko;
  v16h b;
#pragma unroll
  for (int j = 0; j < 16; ++j) b[j] = p[j];
  return b;
}

// Order-preserving float<->uint for atomic segment-max.
DEVI unsigned fkey(float f) {
  unsigned b = __float_as_uint(f);
  return (b & 0x80000000u) ? ~b : (b | 0x80000000u);
}
DEVI float fdec(unsigned u) {
  return (u & 0x80000000u) ? __uint_as_float(u ^ 0x80000000u) : __uint_as_float(~u);
}

// ------------------------------- kernels ------------------------------------

// Pack per-layer weights: Wcat[L][832][64] f16 (transposed), bcat[L][832],
// Wet[L][256][64] f16 (transposed), bet[L][256].
__global__ void k_prep_weights(const float* Wq, const float* bq,
                               const float* Wk, const float* bk,
                               const float* Wv, const float* bv,
                               const float* We, const float* be,
                               const float* Wsk, const float* bsk,
                               _Float16* Wcat, float* bcat,
                               _Float16* Wet, float* bet) {
  const int t_wcat = L_ * WCAT * 64;
  const int t_bcat = L_ * WCAT;
  const int t_wet  = L_ * QKV * 64;
  const int t_bet  = L_ * QKV;
  const int total  = t_wcat + t_bcat + t_wet + t_bet;
  for (int idx = blockIdx.x * blockDim.x + threadIdx.x; idx < total;
       idx += gridDim.x * blockDim.x) {
    int t = idx;
    if (t < t_wcat) {
      int i = t / (WCAT * 64); int r = t % (WCAT * 64);
      int n = r >> 6; int k = r & 63;
      float v;
      if (n < QKV)            v = Wq [(size_t)i * 64 * QKV + k * QKV + n];
      else if (n < 2 * QKV)   v = Wk [(size_t)i * 64 * QKV + k * QKV + (n - QKV)];
      else if (n < 3 * QKV)   v = Wv [(size_t)i * 64 * QKV + k * QKV + (n - 2 * QKV)];
      else                    v = Wsk[(size_t)i * 64 * 64  + k * 64  + (n - 3 * QKV)];
      Wcat[t] = (_Float16)v;
      continue;
    }
    t -= t_wcat;
    if (t < t_bcat) {
      int i = t / WCAT; int n = t % WCAT;
      float v;
      if (n < QKV)          v = bq [i * QKV + n];
      else if (n < 2 * QKV) v = bk [i * QKV + n - QKV];
      else if (n < 3 * QKV) v = bv [i * QKV + n - 2 * QKV];
      else                  v = bsk[i * 64  + n - 3 * QKV];
      bcat[i * WCAT + n] = v;
      continue;
    }
    t -= t_bcat;
    if (t < t_wet) {
      int i = t / (QKV * 64); int r = t % (QKV * 64);
      int n = r >> 6; int k = r & 63;
      Wet[t] = (_Float16)We[(size_t)i * 64 * QKV + k * QKV + n];
      continue;
    }
    t -= t_wet;
    { int i = t / QKV; int n = t % QKV; bet[t] = be[i * QKV + n]; }
  }
}

// h = x @ W_node + b_node  (K=16, memory bound) -> fp32 + f16 copies
__global__ void k_node_encoder(const float* __restrict__ x,
                               const float* __restrict__ Wn, const float* __restrict__ bn,
                               float* hF, _Float16* hB) {
  size_t idx = (size_t)blockIdx.x * blockDim.x + threadIdx.x;
  if (idx >= (size_t)N_ * C_) return;
  int n = (int)(idx >> 6), c = (int)(idx & 63);
  float s = bn[c];
#pragma unroll
  for (int k = 0; k < XD_; ++k) s += x[(size_t)n * XD_ + k] * Wn[k * C_ + c];
  hF[idx] = s;
  hB[idx] = (_Float16)s;
}

// e = edge_attr @ W_edge + b_edge  (K=8) -> f16
__global__ void k_edge_encoder(const float* __restrict__ ea,
                               const float* __restrict__ We, const float* __restrict__ be,
                               _Float16* eB) {
  size_t idx = (size_t)blockIdx.x * blockDim.x + threadIdx.x;
  if (idx >= (size_t)E_ * C_) return;
  int e = (int)(idx >> 6), c = (int)(idx & 63);
  float s = be[c];
#pragma unroll
  for (int k = 0; k < ED_; ++k) s += ea[(size_t)e * ED_ + k] * We[k * C_ + c];
  eB[idx] = (_Float16)s;
}

// Zero per-layer accumulators.
__global__ void k_layer_init(float* agg, float* ssum, unsigned* mmax) {
  const size_t tot = (size_t)N_ * QKV;
  for (size_t idx = (size_t)blockIdx.x * blockDim.x + threadIdx.x; idx < tot;
       idx += (size_t)gridDim.x * blockDim.x) {
    agg[idx] = 0.0f;
    if (idx < (size_t)N_ * H_) { ssum[idx] = 0.0f; mmax[idx] = 0u; }
  }
}

// Fused [Q|K|V|skip] = h @ Wcat + bcat.   grid=(N/16, 13), block=128 (4 waves).
__global__ void k_qkvs_gemm(const _Float16* __restrict__ hB,
                            const _Float16* __restrict__ Wcat,
                            const float* __restrict__ bcat,
                            _Float16* Qh, _Float16* Kh, _Float16* Vh, float* skip) {
  const int lane = threadIdx.x & 31;
  const int nt   = blockIdx.y * 4 + (threadIdx.x >> 5);   // 0..51
  const int mt   = blockIdx.x;                            // 0..2047
  const _Float16* abase = hB + (size_t)mt * 16 * 64;

  v16h a0 = load_a16x32(abase, 64, 0,  lane);
  v16h a1 = load_a16x32(abase, 64, 32, lane);
  v16h b0 = load_b32x16(Wcat, nt * 16, 0,  lane);
  v16h b1 = load_b32x16(Wcat, nt * 16, 32, lane);
  v8f acc = {};
  acc = wmma_f16(a0, b0, acc);
  acc = wmma_f16(a1, b1, acc);

  const int ncol = nt * 16 + (lane & 15);
  const float bias = bcat[ncol];
  const int half = (lane >> 4) << 3;
#pragma unroll
  for (int r = 0; r < 8; ++r) {
    int row = mt * 16 + half + r;
    float v = acc[r] + bias;
    if (nt < 16)      Qh[(size_t)row * QKV + ncol]                = (_Float16)v;
    else if (nt < 32) Kh[(size_t)row * QKV + (ncol - QKV)]        = (_Float16)v;
    else if (nt < 48) Vh[(size_t)row * QKV + (ncol - 2 * QKV)]    = (_Float16)v;
    else              skip[(size_t)row * C_  + (ncol - 3 * QKV)]  = v;
  }
}

// Edge pass 1: ee = e@We (WMMA), alpha[e,h] = scale * q[dst].(k[src]+ee),
// atomic segment max into mmax.   grid = ETILES/4 blocks of 128 threads.
__global__ void k_edge_pass1(const _Float16* __restrict__ eB,
                             const _Float16* __restrict__ Wet,
                             const float* __restrict__ bet,
                             const _Float16* __restrict__ Qh,
                             const _Float16* __restrict__ Kh,
                             const int* __restrict__ ei,
                             float* alpha, unsigned* mmax) {
  __shared__ _Float16 sW[QKV * 64];
  __shared__ float    sb[QKV];
  for (int t = threadIdx.x; t < QKV * 64 / 2; t += blockDim.x)
    ((unsigned*)sW)[t] = ((const unsigned*)Wet)[t];
  for (int t = threadIdx.x; t < QKV; t += blockDim.x) sb[t] = bet[t];
  __syncthreads();

  const int lane = threadIdx.x & 31;
  const int tile = blockIdx.x * 4 + (threadIdx.x >> 5);
  const int e0   = tile * 16;
  const int half = (lane >> 4) << 3;

  const _Float16* abase = eB + (size_t)e0 * 64;
  v16h a0 = load_a16x32(abase, 64, 0,  lane);
  v16h a1 = load_a16x32(abase, 64, 32, lane);

  int srcs[8], dsts[8];
#pragma unroll
  for (int r = 0; r < 8; ++r) {
    srcs[r] = ei[e0 + half + r];
    dsts[r] = ei[E_ + e0 + half + r];
  }

  float part[8] = {0.f, 0.f, 0.f, 0.f, 0.f, 0.f, 0.f, 0.f};
  for (int nt = 0; nt < 16; ++nt) {
    v16h b0 = load_b32x16(sW, nt * 16, 0,  lane);
    v16h b1 = load_b32x16(sW, nt * 16, 32, lane);
    v8f acc = {};
    acc = wmma_f16(a0, b0, acc);
    acc = wmma_f16(a1, b1, acc);

    const int col = nt * 16 + (lane & 15);
    const float bias = sb[col];
#pragma unroll
    for (int r = 0; r < 8; ++r) {
      float kj = acc[r] + bias + (float)Kh[(size_t)srcs[r] * QKV + col];
      part[r] += (float)Qh[(size_t)dsts[r] * QKV + col] * kj;
    }
    if ((nt & 3) == 3) {               // head boundary: reduce 16 lanes of half
      const int head = nt >> 2;
#pragma unroll
      for (int r = 0; r < 8; ++r) {
        float v = part[r];
        v += __shfl_xor(v, 1); v += __shfl_xor(v, 2);
        v += __shfl_xor(v, 4); v += __shfl_xor(v, 8);
        if ((lane & 15) == 0) {
          float av = v * 0.125f;       // 1/sqrt(64)
          alpha[(size_t)(e0 + half + r) * H_ + head] = av;
          atomicMax(&mmax[(size_t)dsts[r] * H_ + head], fkey(av));
        }
        part[r] = 0.f;
      }
    }
  }
}

// Edge pass 2: ex = exp(alpha - m[dst]); ssum[dst,h] += ex;
// agg[dst, :] += ex * (v[src] + ee)  (ee recomputed via WMMA).
__global__ void k_edge_pass2(const _Float16* __restrict__ eB,
                             const _Float16* __restrict__ Wet,
                             const float* __restrict__ bet,
                             const _Float16* __restrict__ Vh,
                             const int* __restrict__ ei,
                             const float* __restrict__ alpha,
                             const unsigned* __restrict__ mmax,
                             float* ssum, float* agg) {
  __shared__ _Float16 sW[QKV * 64];
  __shared__ float    sb[QKV];
  for (int t = threadIdx.x; t < QKV * 64 / 2; t += blockDim.x)
    ((unsigned*)sW)[t] = ((const unsigned*)Wet)[t];
  for (int t = threadIdx.x; t < QKV; t += blockDim.x) sb[t] = bet[t];
  __syncthreads();

  const int lane = threadIdx.x & 31;
  const int tile = blockIdx.x * 4 + (threadIdx.x >> 5);
  const int e0   = tile * 16;
  const int half = (lane >> 4) << 3;

  const _Float16* abase = eB + (size_t)e0 * 64;
  v16h a0 = load_a16x32(abase, 64, 0,  lane);
  v16h a1 = load_a16x32(abase, 64, 32, lane);

  int srcs[8], dsts[8];
#pragma unroll
  for (int r = 0; r < 8; ++r) {
    srcs[r] = ei[e0 + half + r];
    dsts[r] = ei[E_ + e0 + half + r];
  }

  float exh[8];
  for (int nt = 0; nt < 16; ++nt) {
    if ((nt & 3) == 0) {               // new head: refresh ex values
      const int head = nt >> 2;
#pragma unroll
      for (int r = 0; r < 8; ++r) {
        float av = alpha[(size_t)(e0 + half + r) * H_ + head];
        float mv = fdec(mmax[(size_t)dsts[r] * H_ + head]);
        if (!__builtin_isfinite(mv)) mv = 0.f;
        exh[r] = __expf(av - mv);
      }
      if ((lane & 15) == 0) {
#pragma unroll
        for (int r = 0; r < 8; ++r)
          atomicAdd(&ssum[(size_t)dsts[r] * H_ + head], exh[r]);
      }
    }
    v16h b0 = load_b32x16(sW, nt * 16, 0,  lane);
    v16h b1 = load_b32x16(sW, nt * 16, 32, lane);
    v8f acc = {};
    acc = wmma_f16(a0, b0, acc);
    acc = wmma_f16(a1, b1, acc);

    const int col = nt * 16 + (lane & 15);
    const float bias = sb[col];
#pragma unroll
    for (int r = 0; r < 8; ++r) {
      float vj = acc[r] + bias + (float)Vh[(size_t)srcs[r] * QKV + col];
      atomicAdd(&agg[(size_t)dsts[r] * QKV + col], exh[r] * vj);
    }
  }
}

// Normalize + head mean + skip + residual; capture virtual-node embeddings.
__global__ void k_node_finalize(const float* __restrict__ agg,
                                const float* __restrict__ ssum,
                                const float* __restrict__ skip,
                                float* hF, _Float16* hB, float* vemb, int layer) {
  size_t idx = (size_t)blockIdx.x * blockDim.x + threadIdx.x;
  if (idx >= (size_t)N_ * C_) return;
  int n = (int)(idx >> 6), c = (int)(idx & 63);
  float val = 0.f;
#pragma unroll
  for (int h = 0; h < H_; ++h)
    val += agg[(size_t)n * QKV + h * C_ + c] / fmaxf(ssum[(size_t)n * H_ + h], 1e-16f);
  val = val * 0.25f + skip[idx];           // mean over heads + root skip
  float hn = val + hF[idx];                // residual
  if ((n & (NPG - 1)) == NPG - 1)          // virtual node of graph n>>7
    vemb[(size_t)(n >> 7) * (L_ * C_) + layer * C_ + c] = val;
  hF[idx] = hn;
  hB[idx] = (_Float16)hn;
}

// pool = concat(v_emb) @ W_down + b_down ; out = sigmoid(pool @ W_out + b_out)
__global__ void k_pool_out(const float* __restrict__ vemb,
                           const float* __restrict__ Wd, const float* __restrict__ bd,
                           const float* __restrict__ Wo, const float* __restrict__ bo,
                           float* out) {
  const int g = blockIdx.x;
  const int c = threadIdx.x;               // 64 threads
  float s = bd[c];
#pragma unroll 4
  for (int j = 0; j < L_ * C_; ++j)
    s += vemb[(size_t)g * (L_ * C_) + j] * Wd[j * C_ + c];
  __shared__ float red[64];
  red[c] = s * Wo[c];
  __syncthreads();
  for (int off = 32; off > 0; off >>= 1) {
    if (c < off) red[c] += red[c + off];
    __syncthreads();
  }
  if (c == 0) out[g] = 1.f / (1.f + __expf(-(red[0] + bo[0])));
}

// ------------------------------- launcher -----------------------------------

extern "C" void kernel_launch(void* const* d_in, const int* in_sizes, int n_in,
                              void* d_out, int out_size, void* d_ws, size_t ws_size,
                              hipStream_t stream) {
  const float* x    = (const float*)d_in[0];
  const float* ea   = (const float*)d_in[1];
  const int*   ei   = (const int*)d_in[2];
  // d_in[3] = v_idx (implied by (n & 127) == 127)
  const float* Wn   = (const float*)d_in[4];
  const float* bn   = (const float*)d_in[5];
  const float* Wedg = (const float*)d_in[6];
  const float* bedg = (const float*)d_in[7];
  const float* Wq   = (const float*)d_in[8];
  const float* bq   = (const float*)d_in[9];
  const float* Wk   = (const float*)d_in[10];
  const float* bk   = (const float*)d_in[11];
  const float* Wv   = (const float*)d_in[12];
  const float* bv   = (const float*)d_in[13];
  const float* We   = (const float*)d_in[14];
  const float* be   = (const float*)d_in[15];
  const float* Wsk  = (const float*)d_in[16];
  const float* bsk  = (const float*)d_in[17];
  const float* Wd   = (const float*)d_in[18];
  const float* bd   = (const float*)d_in[19];
  const float* Wo   = (const float*)d_in[20];
  const float* bo   = (const float*)d_in[21];
  float* out = (float*)d_out;

  // Workspace carve (all chunks 256B-aligned by construction).
  char* p = (char*)d_ws;
  auto take = [&](size_t bytes) { char* r = p; p += (bytes + 255) & ~(size_t)255; return r; };
  float*     hF    = (float*)    take((size_t)N_ * C_ * 4);
  _Float16*  hB    = (_Float16*) take((size_t)N_ * C_ * 2);
  _Float16*  eBuf  = (_Float16*) take((size_t)E_ * C_ * 2);
  _Float16*  Qh    = (_Float16*) take((size_t)N_ * QKV * 2);
  _Float16*  Kh    = (_Float16*) take((size_t)N_ * QKV * 2);
  _Float16*  Vh    = (_Float16*) take((size_t)N_ * QKV * 2);
  float*     skip  = (float*)    take((size_t)N_ * C_ * 4);
  float*     alpha = (float*)    take((size_t)E_ * H_ * 4);
  unsigned*  mmax  = (unsigned*) take((size_t)N_ * H_ * 4);
  float*     ssum  = (float*)    take((size_t)N_ * H_ * 4);
  float*     agg   = (float*)    take((size_t)N_ * QKV * 4);
  float*     vemb  = (float*)    take((size_t)B_ * L_ * C_ * 4);
  _Float16*  Wcat  = (_Float16*) take((size_t)L_ * WCAT * 64 * 2);
  float*     bcat  = (float*)    take((size_t)L_ * WCAT * 4);
  _Float16*  Wet   = (_Float16*) take((size_t)L_ * QKV * 64 * 2);
  float*     bet   = (float*)    take((size_t)L_ * QKV * 4);

  k_prep_weights<<<1105, 256, 0, stream>>>(Wq, bq, Wk, bk, Wv, bv, We, be, Wsk, bsk,
                                           Wcat, bcat, Wet, bet);
  k_node_encoder<<<(N_ * C_) / 256, 256, 0, stream>>>(x, Wn, bn, hF, hB);
  k_edge_encoder<<<(E_ * C_) / 256, 256, 0, stream>>>(ea, Wedg, bedg, eBuf);

  for (int i = 0; i < L_; ++i) {
    k_layer_init<<<4096, 256, 0, stream>>>(agg, ssum, mmax);
    k_qkvs_gemm<<<dim3(N_ / 16, 13), 128, 0, stream>>>(
        hB, Wcat + (size_t)i * WCAT * 64, bcat + (size_t)i * WCAT, Qh, Kh, Vh, skip);
    k_edge_pass1<<<ETILES / 4, 128, 0, stream>>>(
        eBuf, Wet + (size_t)i * QKV * 64, bet + (size_t)i * QKV,
        Qh, Kh, ei, alpha, mmax);
    k_edge_pass2<<<ETILES / 4, 128, 0, stream>>>(
        eBuf, Wet + (size_t)i * QKV * 64, bet + (size_t)i * QKV,
        Vh, ei, alpha, mmax, ssum, agg);
    k_node_finalize<<<(N_ * C_) / 256, 256, 0, stream>>>(
        agg, ssum, skip, hF, hB, vemb, i);
  }

  k_pool_out<<<B_, 64, 0, stream>>>(vemb, Wd, bd, Wo, bo, out);
}
